// Attention_3375844295055
// MI455X (gfx1250) — matrix operand order repaired and checked by reading, not compiled, not run on hardware
//
#include <hip/hip_runtime.h>

// CDNA5 / gfx1250 wave32 WMMA attention, double-buffered + software pipelined.
// All matmuls via v_wmma_f32_16x16x32_bf16 (fp32 accum, bf16 operands).

typedef __attribute__((ext_vector_type(16))) __bf16 v16bf;
typedef __attribute__((ext_vector_type(8)))  float  v8f;
typedef int v4i __attribute__((vector_size(16)));   // matches async-LDS builtin param

union FragA  { v16bf v; uint4 q[2]; };          // 32 bytes: A or B operand fragment
union FragC  { v8f   v; float f[8]; };          // 32 bytes: C/D accumulator fragment

#ifndef __has_builtin
#define __has_builtin(x) 0
#endif
#if __has_builtin(__builtin_amdgcn_global_load_async_to_lds_b128) && \
    __has_builtin(__builtin_amdgcn_s_wait_asynccnt)
#define USE_ASYNC_LDS 1
#else
#define USE_ASYNC_LDS 0
#endif
#define AS1 __attribute__((address_space(1)))
#define AS3 __attribute__((address_space(3)))

__device__ __forceinline__ unsigned short f2bf(float x) {
    unsigned int u = __float_as_uint(x);
    u += 0x7FFFu + ((u >> 16) & 1u);            // round-to-nearest-even
    return (unsigned short)(u >> 16);
}

// ---------------------------------------------------------------------------
// Tiled GEMM: C[M,N] = A[M,K] * B[K,N] * scale
// A: fp32 or bf16 (template), B: fp32, C: fp32 or bf16 (template).
// Tile: BM=128, BN=128, BK=32. 256 threads = 8 waves, wave grid 4(M) x 2(N),
// each wave computes 32x64 via 2x4 WMMA 16x16x32 tiles (8 wmma / 12 ds_load).
// Double-buffered LDS, global prefetch-to-regs one tile ahead, L2 prefetch
// two tiles ahead.
// ---------------------------------------------------------------------------
#define BM 128
#define BN 128
#define BK 32
#define APITCH 40   // bf16 elems; row stride 80B = 16*5 -> 16B-aligned frag reads
#define BPITCH 40

template <bool A_BF16, bool OUT_BF16>
__global__ __launch_bounds__(256) void gemm_wmma(
    const void* __restrict__ Ag_, const float* __restrict__ Bg,
    void* __restrict__ Cg_, int M, int Nn, int K, float scale)
{
    __shared__ unsigned short Asm[2][BM * APITCH];   // A tile, bf16, row-major
    __shared__ unsigned short Btsm[2][BN * BPITCH];  // B tile, bf16, TRANSPOSED [n][k]

    const int tid  = threadIdx.x;
    const int lane = tid & 31;
    const int wid  = tid >> 5;          // 0..7
    const int wm   = wid >> 1;          // 0..3  (M direction)
    const int wn   = wid & 1;           // 0..1  (N direction)
    const int half = (lane >> 4) & 1;
    const int nl   = lane & 15;

    const int rowBlk = blockIdx.y * BM;
    const int colBlk = blockIdx.x * BN;

    // staging assignments
    const int aC = (tid & 7) * 4;       // A col in BK
    const int aR = tid >> 3;            // A row base 0..31 (+32*i)
    const int bC = (tid & 31) * 4;      // B col in BN
    const int bR = tid >> 5;            // B row base 0..7 (+8*i)

    const v8f vzero = {0.f,0.f,0.f,0.f,0.f,0.f,0.f,0.f};
    v8f acc[2][4];
    for (int ti = 0; ti < 2; ++ti)
        for (int tj = 0; tj < 4; ++tj) acc[ti][tj] = vzero;

    float4 pa[4]; uint2 pa2[4]; float4 pb[4];

    auto prefetchG = [&](int k0) {
#pragma unroll
        for (int i = 0; i < 4; ++i) {
            const int r = aR + i * 32;
            if (A_BF16) {
                const unsigned short* Ag = (const unsigned short*)Ag_;
                pa2[i] = *(const uint2*)(Ag + (size_t)(rowBlk + r) * K + k0 + aC);
            } else {
                const float* Ag = (const float*)Ag_;
                pa[i] = *(const float4*)(Ag + (size_t)(rowBlk + r) * K + k0 + aC);
            }
        }
#pragma unroll
        for (int i = 0; i < 4; ++i) {
            const int r = bR + i * 8;
            pb[i] = *(const float4*)(Bg + (size_t)(k0 + r) * Nn + colBlk + bC);
        }
    };
    auto storeL = [&](int buf) {
#pragma unroll
        for (int i = 0; i < 4; ++i) {
            const int r = aR + i * 32;
            uint2 d;
            if (A_BF16) {
                d = pa2[i];
            } else {
                d.x = (unsigned)f2bf(pa[i].x) | ((unsigned)f2bf(pa[i].y) << 16);
                d.y = (unsigned)f2bf(pa[i].z) | ((unsigned)f2bf(pa[i].w) << 16);
            }
            *(uint2*)(Asm[buf] + r * APITCH + aC) = d;
        }
#pragma unroll
        for (int i = 0; i < 4; ++i) {
            const int r = bR + i * 8;
            Btsm[buf][(bC + 0) * BPITCH + r] = f2bf(pb[i].x);
            Btsm[buf][(bC + 1) * BPITCH + r] = f2bf(pb[i].y);
            Btsm[buf][(bC + 2) * BPITCH + r] = f2bf(pb[i].z);
            Btsm[buf][(bC + 3) * BPITCH + r] = f2bf(pb[i].w);
        }
    };

    prefetchG(0);
    storeL(0);
    __syncthreads();

    const int nK = K / BK;
    for (int it = 0; it < nK; ++it) {
        const int buf = it & 1;
        if (it + 1 < nK) prefetchG((it + 1) * BK);
        if (it + 2 < nK) {                       // warm L2 two tiles ahead
            const int k2 = (it + 2) * BK;
            if (A_BF16)
                __builtin_prefetch((const unsigned short*)Ag_ + (size_t)(rowBlk + aR) * K + k2 + aC, 0, 3);
            else
                __builtin_prefetch((const float*)Ag_ + (size_t)(rowBlk + aR) * K + k2 + aC, 0, 3);
            __builtin_prefetch(Bg + (size_t)(k2 + bR) * Nn + colBlk + bC, 0, 3);
        }

        // ---- fragments ----
        // A 16x32 bf16: lane m=nl, e<8 -> k=half*8+e ; e>=8 -> k=16+half*8+(e-8)
        FragA a[2], bfr[4];
#pragma unroll
        for (int ti = 0; ti < 2; ++ti) {
            const unsigned short* p = Asm[buf] + (wm * 32 + ti * 16 + nl) * APITCH;
            a[ti].q[0] = *(const uint4*)(p + half * 8);
            a[ti].q[1] = *(const uint4*)(p + 16 + half * 8);
        }
        // B 32x16 bf16: lane n=nl, e<8 -> k=half*8+e ; e>=8 -> k=16+half*8+(e-8)
#pragma unroll
        for (int tj = 0; tj < 4; ++tj) {
            const unsigned short* p = Btsm[buf] + (wn * 64 + tj * 16 + nl) * BPITCH + half * 8;
            bfr[tj].q[0] = *(const uint4*)(p);
            bfr[tj].q[1] = *(const uint4*)(p + 16);
        }
#pragma unroll
        for (int ti = 0; ti < 2; ++ti)
#pragma unroll
            for (int tj = 0; tj < 4; ++tj)
                acc[ti][tj] = __builtin_amdgcn_wmma_f32_16x16x32_bf16(
                    false, a[ti].v, false, bfr[tj].v, (short)0, acc[ti][tj], false, false);

        if (it + 1 < nK) storeL(buf ^ 1);
        __syncthreads();
    }

    // ---- epilogue: C layout lane n=nl, elem e -> row m = e + half*8 ----
    for (int ti = 0; ti < 2; ++ti)
        for (int tj = 0; tj < 4; ++tj) {
            FragC r; r.v = acc[ti][tj];
            const int col = colBlk + wn * 64 + tj * 16 + nl;
            for (int e = 0; e < 8; ++e) {
                const int row = rowBlk + wm * 32 + ti * 16 + half * 8 + e;
                const float val = r.f[e] * scale;
                if (OUT_BF16)
                    ((unsigned short*)Cg_)[(size_t)row * Nn + col] = f2bf(val);
                else
                    ((float*)Cg_)[(size_t)row * Nn + col] = val;
            }
        }
}

// ---------------------------------------------------------------------------
// Flash-attention kernel. Q/K/V bf16 [B*L, N*D] (Q pre-scaled by D^-0.5).
// Block: 128 threads = 4 waves, each wave owns 16 f-rows; block tile = 64 f.
// T swept in 64-wide tiles; double-buffered K/V staging pipelined against
// compute. K tile is a straight global->LDS copy: routed through the CDNA5
// async-to-LDS path (ASYNCcnt) when the builtin exists; V needs a transpose
// so it goes global->VGPR->LDS.
// ---------------------------------------------------------------------------
#define KP 72   // LDS pitch in bf16 elems; 144B = 16*9 -> 16B-aligned frag reads

__global__ __launch_bounds__(128) void attn_wmma(
    const unsigned short* __restrict__ Qg,
    const unsigned short* __restrict__ Kg,
    const unsigned short* __restrict__ Vg,
    const float* __restrict__ bias,
    unsigned short* __restrict__ Ctx)
{
    __shared__ unsigned short Klds[2][64 * KP];    // K tile [t][d], double-buffered
    __shared__ unsigned short Vt[2][64 * KP];      // V tile transposed [d][t]
    __shared__ unsigned short Plds[4 * 16 * KP];   // per-wave P scratch [m][t]

    const int tid  = threadIdx.x;
    const int lane = tid & 31;
    const int wid  = tid >> 5;          // 0..3
    const int half = (lane >> 4) & 1;
    const int nl   = lane & 15;

    const int fBlk = blockIdx.x * 64;
    const int head = blockIdx.y;        // 0..15
    const int b    = blockIdx.z;

    // Q fragments for this wave's 16 rows (constant over the T loop)
    FragA aq[2];
    {
        const unsigned short* qr =
            Qg + ((size_t)b * 2048 + fBlk + wid * 16 + nl) * 1024 + head * 64;
#pragma unroll
        for (int kk = 0; kk < 2; ++kk) {
            aq[kk].q[0] = *(const uint4*)(qr + kk * 32 + half * 8);
            aq[kk].q[1] = *(const uint4*)(qr + kk * 32 + 16 + half * 8);
        }
    }

    const v8f vzero = {0.f,0.f,0.f,0.f,0.f,0.f,0.f,0.f};
    FragC o[4];
    for (int dj = 0; dj < 4; ++dj) o[dj].v = vzero;
    float rm[8], rl[8];
    for (int e = 0; e < 8; ++e) { rm[e] = -1.0e30f; rl[e] = 0.0f; }

    unsigned short* Pw = Plds + wid * 16 * KP;

    uint4 vreg[4], kreg[4];
    (void)kreg;

    // issue global loads for tile at `tt` targeting LDS buffer `dbuf`
    auto issueTile = [&](int tt, int dbuf) {
#pragma unroll
        for (int i = 0; i < 4; ++i) {
            const int seg = tid + i * 128;          // 512 segments of 8 bf16
            const int r   = seg >> 3;               // t row 0..63
            const int off = (seg & 7) << 3;         // d offset 0..56
            const size_t gbase = ((size_t)b * 2048 + tt + r) * 1024 + head * 64 + off;
#if USE_ASYNC_LDS
            __builtin_amdgcn_global_load_async_to_lds_b128(
                (AS1 v4i*)(unsigned long long)(Kg + gbase),
                (AS3 v4i*)(unsigned int)(unsigned long long)(Klds[dbuf] + r * KP + off),
                0, 0);
#else
            kreg[i] = *(const uint4*)(Kg + gbase);
#endif
            vreg[i] = *(const uint4*)(Vg + gbase);
        }
    };
    // finish staging: V transpose from regs (and K from regs in fallback path)
    auto storeTile = [&](int dbuf) {
#pragma unroll
        for (int i = 0; i < 4; ++i) {
            const int seg = tid + i * 128;
            const int r   = seg >> 3;
            const int off = (seg & 7) << 3;
#if !USE_ASYNC_LDS
            *(uint4*)(Klds[dbuf] + r * KP + off) = kreg[i];
#endif
            const uint4 vq = vreg[i];
            Vt[dbuf][(off + 0) * KP + r] = (unsigned short)(vq.x);
            Vt[dbuf][(off + 1) * KP + r] = (unsigned short)(vq.x >> 16);
            Vt[dbuf][(off + 2) * KP + r] = (unsigned short)(vq.y);
            Vt[dbuf][(off + 3) * KP + r] = (unsigned short)(vq.y >> 16);
            Vt[dbuf][(off + 4) * KP + r] = (unsigned short)(vq.z);
            Vt[dbuf][(off + 5) * KP + r] = (unsigned short)(vq.z >> 16);
            Vt[dbuf][(off + 6) * KP + r] = (unsigned short)(vq.w);
            Vt[dbuf][(off + 7) * KP + r] = (unsigned short)(vq.w >> 16);
        }
    };

    issueTile(0, 0);
    storeTile(0);
#if USE_ASYNC_LDS
    __builtin_amdgcn_s_wait_asynccnt(0);
#endif
    __syncthreads();

    for (int it = 0; it < 32; ++it) {
        const int tt  = it * 64;
        const int buf = it & 1;
        const bool hasNext = (it + 1) < 32;
        // issue next tile's global traffic first (overlaps with all compute below;
        // WAR on Klds[buf^1]/Vt[buf^1] is safe: last read at barrier of it-1)
        if (hasNext) issueTile(tt + 64, buf ^ 1);

        // ---- S = Q * K^T : 4 tiles of 16f x 16t ----
        FragC s[4];
#pragma unroll
        for (int tj = 0; tj < 4; ++tj) {
            v8f acc = vzero;
#pragma unroll
            for (int kk = 0; kk < 2; ++kk) {
                FragA bk;   // B elem: K[t = tj*16+nl][d = kk*32 + table-order k(half,e)]
                const unsigned short* p = Klds[buf] + (tj * 16 + nl) * KP + kk * 32 + half * 8;
                bk.q[0] = *(const uint4*)(p);
                bk.q[1] = *(const uint4*)(p + 16);
                acc = __builtin_amdgcn_wmma_f32_16x16x32_bf16(
                    false, aq[kk].v, false, bk.v, (short)0, acc, false, false);
            }
            s[tj].v = acc;
        }

        // ---- + bias[b,0,f,t] (fits in 192MB L2; reused across the 16 heads) ----
#pragma unroll
        for (int tj = 0; tj < 4; ++tj)
            for (int e = 0; e < 8; ++e) {
                const int f = fBlk + wid * 16 + half * 8 + e;
                const int t = tt + tj * 16 + nl;
                s[tj].f[e] += bias[((size_t)b * 2048 + f) * 2048 + t];
            }

        // ---- online softmax: row reductions across the 16-lane N group ----
        float corr[8];
        for (int e = 0; e < 8; ++e) {
            float mx = fmaxf(fmaxf(s[0].f[e], s[1].f[e]), fmaxf(s[2].f[e], s[3].f[e]));
            for (int off = 8; off >= 1; off >>= 1)
                mx = fmaxf(mx, __shfl_xor(mx, off, 32));
            const float nm = fmaxf(rm[e], mx);
            corr[e] = __expf(rm[e] - nm);
            rm[e] = nm;
        }
        for (int e = 0; e < 8; ++e) {
            float sum = 0.0f;
            for (int tj = 0; tj < 4; ++tj) {
                const float p = __expf(s[tj].f[e] - rm[e]);
                s[tj].f[e] = p;
                sum += p;
            }
            for (int off = 8; off >= 1; off >>= 1)
                sum += __shfl_xor(sum, off, 32);
            rl[e] = rl[e] * corr[e] + sum;
        }
        for (int dj = 0; dj < 4; ++dj)
            for (int e = 0; e < 8; ++e)
                o[dj].f[e] *= corr[e];

        // ---- P -> per-wave LDS (bf16), then re-read as A fragments ----
        // (wave-private region: per-wave LDS ordering suffices, no barrier)
#pragma unroll
        for (int tj = 0; tj < 4; ++tj)
            for (int e = 0; e < 8; ++e)
                Pw[(half * 8 + e) * KP + tj * 16 + nl] = f2bf(s[tj].f[e]);

        // ---- O += P * V ----
#pragma unroll
        for (int kt = 0; kt < 2; ++kt) {
            FragA ap;
            const unsigned short* pp = Pw + nl * KP + kt * 32;
            ap.q[0] = *(const uint4*)(pp + half * 8);
            ap.q[1] = *(const uint4*)(pp + 16 + half * 8);
#pragma unroll
            for (int dj = 0; dj < 4; ++dj) {
                FragA bv;   // B elem: V[t = kt*32 + table-order k(half,e)][d = dj*16 + nl]
                const unsigned short* vp = Vt[buf] + (dj * 16 + nl) * KP + kt * 32 + half * 8;
                bv.q[0] = *(const uint4*)(vp);
                bv.q[1] = *(const uint4*)(vp + 16);
                o[dj].v = __builtin_amdgcn_wmma_f32_16x16x32_bf16(
                    false, ap.v, false, bv.v, (short)0, o[dj].v, false, false);
            }
        }

        // finish staging next tile into the alternate buffers
        if (hasNext) storeTile(buf ^ 1);
#if USE_ASYNC_LDS
        __builtin_amdgcn_s_wait_asynccnt(0);
#endif
        __syncthreads();
    }

    // ---- normalize + store ctx bf16 [B*F, N*D] ----
    for (int e = 0; e < 8; ++e) rl[e] = 1.0f / rl[e];
    for (int dj = 0; dj < 4; ++dj)
        for (int e = 0; e < 8; ++e) {
            const int f = fBlk + wid * 16 + half * 8 + e;
            const int d = head * 64 + dj * 16 + nl;
            Ctx[((size_t)b * 2048 + f) * 1024 + d] = f2bf(o[dj].f[e] * rl[e]);
        }
}

// ---------------------------------------------------------------------------
// B=2, F=T=2048, H=1024, N=16, D=64.
// Workspace: Q, K, V, Ctx as bf16 [4096, 1024] -> 4 x 8 MiB = 32 MiB.
// ---------------------------------------------------------------------------
extern "C" void kernel_launch(void* const* d_in, const int* in_sizes, int n_in,
                              void* d_out, int out_size, void* d_ws, size_t ws_size,
                              hipStream_t stream) {
    const float* q_in = (const float*)d_in[0];   // [2,2048,1024]
    const float* s_in = (const float*)d_in[1];   // [2,2048,1024]
    const float* bias = (const float*)d_in[2];   // [2,1,2048,2048]
    const float* wq   = (const float*)d_in[3];   // [1024,16,64] = [1024,1024]
    const float* wk   = (const float*)d_in[4];
    const float* wv   = (const float*)d_in[5];
    const float* wo   = (const float*)d_in[6];   // [16,64,1024] = [1024,1024]
    float* out = (float*)d_out;                  // [2,2048,1024]

    unsigned short* qws = (unsigned short*)d_ws;
    unsigned short* kws = qws + (size_t)4096 * 1024;
    unsigned short* vws = kws + (size_t)4096 * 1024;
    unsigned short* cws = vws + (size_t)4096 * 1024;

    const dim3 gg(1024 / BN, 4096 / BM);   // (8, 32)
    // Q projection absorbs D^-0.5 = 0.125
    gemm_wmma<false, true><<<gg, 256, 0, stream>>>(q_in, wq, qws, 4096, 1024, 1024, 0.125f);
    gemm_wmma<false, true><<<gg, 256, 0, stream>>>(s_in, wk, kws, 4096, 1024, 1024, 1.0f);
    gemm_wmma<false, true><<<gg, 256, 0, stream>>>(s_in, wv, vws, 4096, 1024, 1024, 1.0f);

    const dim3 ga(2048 / 64, 16, 2);       // (f-tiles, heads, batch)
    attn_wmma<<<ga, 128, 0, stream>>>(qws, kws, vws, bias, cws);

    gemm_wmma<true, false><<<gg, 256, 0, stream>>>(cws, wo, out, 4096, 1024, 1024, 1.0f);
}
